// ReferenceCompilerCompatRouter_13443247636823
// MI455X (gfx1250) — matrix operand order repaired and checked by reading, not compiled, not run on hardware
//
#include <hip/hip_runtime.h>
#include <hip/hip_bf16.h>

#define HIDDEN      7168
#define NUM_EXPERTS 256
#define TOKENS      8192
#define TOPK        8
#define NGROUP      8
#define EPG         32
#define TOPKG       4

typedef __attribute__((ext_vector_type(16))) __bf16 v16bf;
typedef __attribute__((ext_vector_type(8)))  float  v8f;
typedef __attribute__((ext_vector_type(4)))  float  f32x4;

// ---------------------------------------------------------------------------
// Kernel 1: one-time split of the fp32 router weight into bf16 hi + lo parts.
// W is only 7.3 MB; cost is negligible and it keeps the B-side split out of the GEMM loop.
// ---------------------------------------------------------------------------
__global__ __launch_bounds__(256) void convert_w(const float* __restrict__ w,
                                                 __bf16* __restrict__ hi,
                                                 __bf16* __restrict__ lo) {
    size_t i = ((size_t)blockIdx.x * blockDim.x + threadIdx.x) * 4;
    f32x4 v = *(const f32x4*)(w + i);
#pragma unroll
    for (int e = 0; e < 4; ++e) {
        float  f = v[e];
        __bf16 h = (__bf16)f;
        hi[i + e] = h;
        lo[i + e] = (__bf16)(f - (float)h);
    }
}

// ---------------------------------------------------------------------------
// Kernel 2: router GEMM with bf16x3 error-compensated WMMA + fused sigmoid.
// Block = 256 threads = 8 waves. Block owns 16 tokens x 256 experts.
// Wave w owns 16 tokens x experts [32w, 32w+32): two 16x16 tiles.
// A (x) is loaded f32 and split hi/lo on the fly (v_cvt_pk_bf16_f32 +
// v_fma_mix_f32_bf16 co-execute with the XDL matrix pipe).
// K unrolled x2 (64/trip); x stream prefetched ~16 K-steps ahead.
// ---------------------------------------------------------------------------
__global__ __launch_bounds__(256) void router_gemm(const float*  __restrict__ x,
                                                   const __bf16* __restrict__ wHi,
                                                   const __bf16* __restrict__ wLo,
                                                   float* __restrict__ scores) {
    const int lane    = threadIdx.x & 31;
    const int wave    = threadIdx.x >> 5;
    const int tok0    = blockIdx.x * 16;
    const int n0      = wave * 32;
    const int m       = lane & 15;
    const int halfsel = lane >> 4;

    // A k order used here (contiguous, matches B): lanes 0-15 K = 0..15; lanes 16-31 K = 16..31.
    const float* xrow = x + (size_t)(tok0 + m) * HIDDEN + 16 * halfsel;
    // B k order used here (contiguous, matches A): lanes 0-15 K = 0..15; lanes 16-31 K = 16..31.
    const __bf16* b0h = wHi + (size_t)(n0 + m)      * HIDDEN + 16 * halfsel;
    const __bf16* b0l = wLo + (size_t)(n0 + m)      * HIDDEN + 16 * halfsel;
    const __bf16* b1h = wHi + (size_t)(n0 + 16 + m) * HIDDEN + 16 * halfsel;
    const __bf16* b1l = wLo + (size_t)(n0 + 16 + m) * HIDDEN + 16 * halfsel;

    v8f acc0 = {};
    v8f acc1 = {};

    for (int k0 = 0; k0 < HIDDEN; k0 += 64) {
        // Keep the 235 MB x stream ahead of the load clauses (global_prefetch_b8).
        if (k0 + 512 < HIDDEN) {
            __builtin_prefetch(xrow + k0 + 512, 0, 0);
        }
#pragma unroll
        for (int ku = 0; ku < 2; ++ku) {
            const int k = k0 + 32 * ku;

            f32x4 a0 = *(const f32x4*)(xrow + k + 0);
            f32x4 a1 = *(const f32x4*)(xrow + k + 4);
            f32x4 a2 = *(const f32x4*)(xrow + k + 8);
            f32x4 a3 = *(const f32x4*)(xrow + k + 12);

            float af[16];
#pragma unroll
            for (int e = 0; e < 4; ++e) {
                af[e]      = a0[e];
                af[e + 4]  = a1[e];
                af[e + 8]  = a2[e];
                af[e + 12] = a3[e];
            }
            v16bf ahi, alo;
#pragma unroll
            for (int e = 0; e < 16; ++e) {
                float  f = af[e];
                __bf16 h = (__bf16)f;
                ahi[e] = h;
                alo[e] = (__bf16)(f - (float)h);
            }

            v16bf bh0 = *(const v16bf*)(b0h + k);
            v16bf bl0 = *(const v16bf*)(b0l + k);
            v16bf bh1 = *(const v16bf*)(b1h + k);
            v16bf bl1 = *(const v16bf*)(b1l + k);

            // bf16x3: hi*hi + lo*hi + hi*lo (lo*lo is below the fp32 noise floor)
            acc0 = __builtin_amdgcn_wmma_f32_16x16x32_bf16(false, ahi, false, bh0,
                                                           (short)0, acc0, false, false);
            acc0 = __builtin_amdgcn_wmma_f32_16x16x32_bf16(false, alo, false, bh0,
                                                           (short)0, acc0, false, false);
            acc0 = __builtin_amdgcn_wmma_f32_16x16x32_bf16(false, ahi, false, bl0,
                                                           (short)0, acc0, false, false);

            acc1 = __builtin_amdgcn_wmma_f32_16x16x32_bf16(false, ahi, false, bh1,
                                                           (short)0, acc1, false, false);
            acc1 = __builtin_amdgcn_wmma_f32_16x16x32_bf16(false, alo, false, bh1,
                                                           (short)0, acc1, false, false);
            acc1 = __builtin_amdgcn_wmma_f32_16x16x32_bf16(false, ahi, false, bl1,
                                                           (short)0, acc1, false, false);
        }
    }

    // C/D layout: VGPR r -> M = r + 8*halfsel, N = lane & 15. Fused sigmoid.
    const int mOut = 8 * halfsel;
    const int nCol = lane & 15;
#pragma unroll
    for (int r = 0; r < 8; ++r) {
        float s0 = 1.0f / (1.0f + __expf(-acc0[r]));
        float s1 = 1.0f / (1.0f + __expf(-acc1[r]));
        size_t row = (size_t)(tok0 + mOut + r) * NUM_EXPERTS;
        scores[row + n0 + nCol]      = s0;
        scores[row + n0 + 16 + nCol] = s1;
    }
}

// ---------------------------------------------------------------------------
// Kernel 3: per-token grouped top-k. One wave32 per token; lane L of group j
// holds expert j*32+L. Group sums by shfl reductions, rank-select top-4
// groups, 8 wave-argmax passes for top-8, normalize, scale.
// ---------------------------------------------------------------------------
__global__ __launch_bounds__(256) void router_topk(const float* __restrict__ scores,
                                                   const float* __restrict__ bias,
                                                   int*   __restrict__ outIdx,
                                                   float* __restrict__ outW) {
    const int lane = threadIdx.x & 31;
    const int wave = threadIdx.x >> 5;
    const int t    = blockIdx.x * 8 + wave;
    const float* srow = scores + (size_t)t * NUM_EXPERTS;

    float s[NGROUP], sc[NGROUP], gs[NGROUP];
#pragma unroll
    for (int j = 0; j < NGROUP; ++j) {
        s[j]  = srow[j * EPG + lane];
        sc[j] = s[j] + bias[j * EPG + lane];
        float v = sc[j];
#pragma unroll
        for (int msk = 16; msk >= 1; msk >>= 1) v += __shfl_xor(v, msk, 32);
        gs[j] = v;
    }

    // top-4 groups by rank (tie -> lower group index, matching lax.top_k)
    float cand[NGROUP];
#pragma unroll
    for (int j = 0; j < NGROUP; ++j) {
        int beats = 0;
#pragma unroll
        for (int g = 0; g < NGROUP; ++g) {
            if (g == j) continue;
            if (gs[g] > gs[j] || (gs[g] == gs[j] && g < j)) beats++;
        }
        cand[j] = (beats < TOPKG) ? sc[j] : -3.0e38f;
    }

    float myW = 0.0f, wsum = 0.0f;
    int   myIdx = 0;
    for (int it = 0; it < TOPK; ++it) {
        float lv = -3.0e38f, lw = 0.0f;
        int   li = 0;
#pragma unroll
        for (int j = 0; j < NGROUP; ++j)
            if (cand[j] > lv) { lv = cand[j]; li = j; lw = s[j]; }
        int le = li * EPG + lane;
        // wave argmax carrying (value, expert, raw sigmoid weight)
#pragma unroll
        for (int msk = 16; msk >= 1; msk >>= 1) {
            float ov = __shfl_xor(lv, msk, 32);
            int   oe = __shfl_xor(le, msk, 32);
            float ow = __shfl_xor(lw, msk, 32);
            if (ov > lv || (ov == lv && oe < le)) { lv = ov; le = oe; lw = ow; }
        }
        if (lane == (le & 31)) {
#pragma unroll
            for (int j = 0; j < NGROUP; ++j)
                if (j == (le >> 5)) cand[j] = -3.0e38f;
        }
        if (lane == it) { myIdx = le; myW = lw; }
        wsum += lw;
    }

    float scale = 2.5f / (wsum + 1e-20f);
    if (lane < TOPK) {
        outIdx[(size_t)t * TOPK + lane] = myIdx;
        outW[(size_t)t * TOPK + lane]   = myW * scale;
    }
}

// ---------------------------------------------------------------------------
extern "C" void kernel_launch(void* const* d_in, const int* in_sizes, int n_in,
                              void* d_out, int out_size, void* d_ws, size_t ws_size,
                              hipStream_t stream) {
    const float* x    = (const float*)d_in[0];
    const float* w    = (const float*)d_in[1];
    const float* bias = (const float*)d_in[2];

    char* ws = (char*)d_ws;
    __bf16* wHi    = (__bf16*)ws;                                       // 3.67 MB
    __bf16* wLo    = (__bf16*)(ws + (size_t)NUM_EXPERTS * HIDDEN * 2);  // 3.67 MB
    float*  scores = (float*) (ws + (size_t)NUM_EXPERTS * HIDDEN * 4);  // 8 MB

    int*   outIdx = (int*)d_out;                              // [T, 8] int32
    float* outW   = (float*)d_out + (size_t)TOKENS * TOPK;    // [T, 8] f32

    convert_w<<<(NUM_EXPERTS * HIDDEN) / (256 * 4), 256, 0, stream>>>(w, wHi, wLo);
    router_gemm<<<TOKENS / 16, 256, 0, stream>>>(x, wHi, wLo, scores);
    router_topk<<<TOKENS / 8, 256, 0, stream>>>(scores, bias, outIdx, outW);
}
